// General_TransformerBlock_13580686590375
// MI455X (gfx1250) — matrix operand order repaired and checked by reading, not compiled, not run on hardware
//
#include <hip/hip_runtime.h>

#define BATCH   2
#define CDIM    192
#define PIX     16384     // 128*128
#define HH      128
#define WW      128
#define NHEADS  6
#define HIDDIM  576

typedef __bf16 bf16_t;
typedef bf16_t v16bf __attribute__((ext_vector_type(16)));
typedef bf16_t v8bf  __attribute__((ext_vector_type(8)));
typedef float  v8f   __attribute__((ext_vector_type(8)));

__device__ __forceinline__ bf16_t f2bf(float f) {
  unsigned u = __builtin_bit_cast(unsigned, f);
  unsigned r = u + 0x7FFFu + ((u >> 16) & 1u);   // round-to-nearest-even
  unsigned short h = (unsigned short)(r >> 16);
  return __builtin_bit_cast(bf16_t, h);
}

// A fragment (16x32 bf16): lane = row m; two contiguous 8-elem chunks at +0,+16
__device__ __forceinline__ v16bf load_frag_a(const bf16_t* p) {
  v8bf lo = *reinterpret_cast<const v8bf*>(p);
  v8bf hi = *reinterpret_cast<const v8bf*>(p + 16);
  return __builtin_shufflevector(lo, hi, 0,1,2,3,4,5,6,7,8,9,10,11,12,13,14,15);
}
// B fragment (32x16 bf16): lane = col n; two contiguous 8-elem chunks at +0,+16
__device__ __forceinline__ v16bf load_frag_b(const bf16_t* p) {
  v8bf lo = *reinterpret_cast<const v8bf*>(p);
  v8bf hi = *reinterpret_cast<const v8bf*>(p + 16);
  return __builtin_shufflevector(lo, hi, 0,1,2,3,4,5,6,7,8,9,10,11,12,13,14,15);
}

// ---------------------------------------------------------------------------
// Generic bf16 WMMA GEMM:  C[m,n] = sum_k A[m,k] * B[n-majorK](k)   (+ C if acc)
//   A: row-major [M,K], lda = row stride (contiguous K)
//   B: "K-contiguous" addressing: element(k,n) at B[n*ldb + k]  (NHWC activations)
//   C: fp32, element(m,n) at C[m*ldc + n] (NCHW)            [optional]
//   Cbf: bf16 NHWC, element at Cbf[n*ldcbf + m]             [optional]
// Per-operand batch offset: off(z) = (z/mod)*outer + (z%mod)*inner
// One wave computes a 32x32 tile (2x2 WMMA subtiles); block = 8 waves
// = 64(M) x 128(N) tile. Each k-step: 8 b128 loads feed 4 WMMAs.
// ---------------------------------------------------------------------------
__global__ __launch_bounds__(256) void gemm_wmma_bf16(
    const bf16_t* __restrict__ A, const bf16_t* __restrict__ Bm,
    float* __restrict__ C, bf16_t* __restrict__ Cbf,
    int M, int N, int K, int lda, int ldb, int ldc, int ldcbf,
    long aO, long aI, int aM,
    long bO, long bI, int bM,
    long cO, long cI, int cM,
    long cbO, long cbI, int cbM,
    int accumulate)
{
  int z = blockIdx.z;
  const bf16_t* Ap = A + (long)(z / aM) * aO + (long)(z % aM) * aI;
  const bf16_t* Bp = Bm + (long)(z / bM) * bO + (long)(z % bM) * bI;
  float* Cp = C ? C + (long)(z / cM) * cO + (long)(z % cM) * cI : nullptr;
  bf16_t* Cbp = Cbf ? Cbf + (long)(z / cbM) * cbO + (long)(z % cbM) * cbI : nullptr;

  int wave = threadIdx.x >> 5;
  int lane = threadIdx.x & 31;
  int m0 = blockIdx.y * 64 + (wave >> 2) * 32;
  int n0 = blockIdx.x * 128 + (wave & 3) * 32;
  if (m0 >= M || n0 >= N) return;

  int l15 = lane & 15;
  int g   = lane >> 4;

  const bf16_t* aRow0 = Ap + (long)(m0 + l15) * lda + g * 8;
  const bf16_t* aRow1 = aRow0 + (long)16 * lda;
  const bf16_t* bRow0 = Bp + (long)(n0 + l15) * ldb + g * 8;
  const bf16_t* bRow1 = bRow0 + (long)16 * ldb;

  v8f acc00 = {}, acc01 = {}, acc10 = {}, acc11 = {};
  if (accumulate && Cp) {
    #pragma unroll
    for (int v = 0; v < 8; ++v) {
      long r0 = (long)(m0 + v + 8 * g) * ldc;
      long r1 = (long)(m0 + 16 + v + 8 * g) * ldc;
      acc00[v] = Cp[r0 + n0 + l15];
      acc01[v] = Cp[r0 + n0 + 16 + l15];
      acc10[v] = Cp[r1 + n0 + l15];
      acc11[v] = Cp[r1 + n0 + 16 + l15];
    }
  }

  for (int k0 = 0; k0 < K; k0 += 32) {
    v16bf a0 = load_frag_a(aRow0 + k0);
    v16bf a1 = load_frag_a(aRow1 + k0);
    v16bf b0 = load_frag_b(bRow0 + k0);
    v16bf b1 = load_frag_b(bRow1 + k0);
    if (k0 + 128 < K) {
      __builtin_prefetch(aRow0 + k0 + 128, 0, 1);
      __builtin_prefetch(aRow1 + k0 + 128, 0, 1);
      __builtin_prefetch(bRow0 + k0 + 128, 0, 1);
      __builtin_prefetch(bRow1 + k0 + 128, 0, 1);
    }
    acc00 = __builtin_amdgcn_wmma_f32_16x16x32_bf16(false, a0, false, b0, (short)0, acc00, false, false);
    acc01 = __builtin_amdgcn_wmma_f32_16x16x32_bf16(false, a0, false, b1, (short)0, acc01, false, false);
    acc10 = __builtin_amdgcn_wmma_f32_16x16x32_bf16(false, a1, false, b0, (short)0, acc10, false, false);
    acc11 = __builtin_amdgcn_wmma_f32_16x16x32_bf16(false, a1, false, b1, (short)0, acc11, false, false);
  }

  if (Cp) {
    #pragma unroll
    for (int v = 0; v < 8; ++v) {
      long r0 = (long)(m0 + v + 8 * g) * ldc;
      long r1 = (long)(m0 + 16 + v + 8 * g) * ldc;
      Cp[r0 + n0 + l15]      = acc00[v];
      Cp[r0 + n0 + 16 + l15] = acc01[v];
      Cp[r1 + n0 + l15]      = acc10[v];
      Cp[r1 + n0 + 16 + l15] = acc11[v];
    }
  }
  if (Cbp) {
    v8bf ob00, ob01, ob10, ob11;
    #pragma unroll
    for (int v = 0; v < 8; ++v) {
      ob00[v] = f2bf(acc00[v]); ob01[v] = f2bf(acc01[v]);
      ob10[v] = f2bf(acc10[v]); ob11[v] = f2bf(acc11[v]);
    }
    long rn0 = (long)(n0 + l15) * ldcbf;
    long rn1 = (long)(n0 + 16 + l15) * ldcbf;
    *reinterpret_cast<v8bf*>(Cbp + rn0 + m0 + 8 * g)      = ob00;
    *reinterpret_cast<v8bf*>(Cbp + rn1 + m0 + 8 * g)      = ob01;
    *reinterpret_cast<v8bf*>(Cbp + rn0 + m0 + 16 + 8 * g) = ob10;
    *reinterpret_cast<v8bf*>(Cbp + rn1 + m0 + 16 + 8 * g) = ob11;
  }
}

// --------------------------- support kernels -------------------------------

__global__ void f32_to_bf16_kernel(const float* __restrict__ in,
                                   bf16_t* __restrict__ out, long n) {
  long i = (long)blockIdx.x * blockDim.x + threadIdx.x;
  if (i < n) out[i] = f2bf(in[i]);
}

// LayerNorm over channel dim at each position; emits fp32 NCHW and/or bf16 NHWC
__global__ void layernorm_kernel(const float* __restrict__ x,
                                 const float* __restrict__ w, const float* __restrict__ b,
                                 float* __restrict__ of32, bf16_t* __restrict__ obf, int Cc)
{
  long idx = (long)blockIdx.x * blockDim.x + threadIdx.x;  // over BATCH*PIX
  if (idx >= (long)BATCH * PIX) return;
  int bi = (int)(idx / PIX), p = (int)(idx % PIX);
  const float* xp = x + (long)bi * Cc * PIX + p;
  float s = 0.f, ss = 0.f;
  for (int c = 0; c < Cc; ++c) { float v = xp[(long)c * PIX]; s += v; ss += v * v; }
  float mu  = s / Cc;
  float var = ss / Cc - mu * mu;
  float inv = rsqrtf(var + 1e-5f);
  bf16_t* ob = obf ? obf + idx * Cc : nullptr;
  for (int c = 0; c < Cc; ++c) {
    float v = (xp[(long)c * PIX] - mu) * inv * w[c] + b[c];
    if (of32) of32[(long)bi * Cc * PIX + (long)c * PIX + p] = v;
    if (ob) ob[c] = f2bf(v);
  }
}

// depthwise 3x3, SAME padding, NCHW fp32
__global__ void dwconv3_kernel(const float* __restrict__ in, const float* __restrict__ w,
                               float* __restrict__ out, int Cc)
{
  long idx = (long)blockIdx.x * blockDim.x + threadIdx.x;
  long total = (long)BATCH * Cc * PIX;
  if (idx >= total) return;
  int p = (int)(idx % PIX);
  long bc = idx / PIX;
  int c = (int)(bc % Cc);
  int y = p >> 7, xx = p & (WW - 1);
  const float* plane = in + (idx - p);
  const float* wp = w + (long)c * 9;
  float acc = 0.f;
  #pragma unroll
  for (int dy = -1; dy <= 1; ++dy) {
    int yy = y + dy;
    if (yy < 0 || yy >= HH) continue;
    #pragma unroll
    for (int dx = -1; dx <= 1; ++dx) {
      int xc = xx + dx;
      if (xc < 0 || xc >= WW) continue;
      acc += plane[yy * WW + xc] * wp[(dy + 1) * 3 + (dx + 1)];
    }
  }
  out[idx] = acc;
}

// L2-normalize each row of length PIX, emit bf16 (NCHW rows)
__global__ void l2norm_rows_kernel(const float* __restrict__ in, long batchStride,
                                   int rowsPerBatch, bf16_t* __restrict__ out)
{
  int row = blockIdx.x;
  int bi = row / rowsPerBatch, c = row % rowsPerBatch;
  const float* ip = in + (long)bi * batchStride + (long)c * PIX;
  __shared__ float red[256];
  float ss = 0.f;
  for (int p = threadIdx.x; p < PIX; p += 256) { float v = ip[p]; ss += v * v; }
  red[threadIdx.x] = ss;
  __syncthreads();
  for (int s = 128; s > 0; s >>= 1) {
    if ((int)threadIdx.x < s) red[threadIdx.x] += red[threadIdx.x + s];
    __syncthreads();
  }
  float inv = 1.f / fmaxf(sqrtf(red[0]), 1e-12f);
  bf16_t* op = out + (long)row * PIX;
  for (int p = threadIdx.x; p < PIX; p += 256) op[p] = f2bf(ip[p] * inv);
}

// NCHW fp32 channel-slice -> NHWC bf16
__global__ void nchw_to_nhwc_bf16_kernel(const float* __restrict__ in, bf16_t* __restrict__ out,
                                         int cOut, int cInTot, int cInOff)
{
  long idx = (long)blockIdx.x * blockDim.x + threadIdx.x;  // (b,p,c)
  long total = (long)BATCH * PIX * cOut;
  if (idx >= total) return;
  int c = (int)(idx % cOut);
  long bp = idx / cOut;
  int p = (int)(bp % PIX), bi = (int)(bp / PIX);
  out[idx] = f2bf(in[(long)bi * cInTot * PIX + (long)(cInOff + c) * PIX + p]);
}

// DDF elementwise: o1 = (c1+1)*x ; o2 = (c2+1)*mt  (bf16 NHWC outputs)
__global__ void ddf_elemwise_kernel(const float* __restrict__ cc,
                                    const float* __restrict__ x, long xBS, int xCOff,
                                    const float* __restrict__ mt,
                                    bf16_t* __restrict__ o1, bf16_t* __restrict__ o2, int d)
{
  long idx = (long)blockIdx.x * blockDim.x + threadIdx.x;  // (b,c,p)
  long total = (long)BATCH * d * PIX;
  if (idx >= total) return;
  int p = (int)(idx % PIX);
  long bc = idx / PIX;
  int c = (int)(bc % d), bi = (int)(bc / d);
  long cb = (long)bi * 2 * d * PIX;
  float c1 = cc[cb + (long)c * PIX + p];
  float c2 = cc[cb + (long)(c + d) * PIX + p];
  float xv = x[(long)bi * xBS + (long)(xCOff + c) * PIX + p];
  float mv = mt[(long)bi * d * PIX + (long)c * PIX + p];
  long ni = ((long)bi * PIX + p) * d + c;
  o1[ni] = f2bf(c1 * xv + xv);
  o2[ni] = f2bf(c2 * mv + mv);
}

// out = a + b_slice ; optional bf16 NHWC copy
__global__ void residual_kernel(const float* __restrict__ a,
                                const float* __restrict__ bsrc, long bBS, int bCOff,
                                float* __restrict__ of32, bf16_t* __restrict__ obf, int d)
{
  long idx = (long)blockIdx.x * blockDim.x + threadIdx.x;  // (b,c,p)
  long total = (long)BATCH * d * PIX;
  if (idx >= total) return;
  int p = (int)(idx % PIX);
  long bc = idx / PIX;
  int c = (int)(bc % d), bi = (int)(bc / d);
  float v = a[idx] + bsrc[(long)bi * bBS + (long)(bCOff + c) * PIX + p];
  if (of32) of32[idx] = v;
  if (obf) obf[((long)bi * PIX + p) * d + c] = f2bf(v);
}

// softmax over 32-wide rows with per-head temperature; bf16 out
__global__ void softmax_kernel(const float* __restrict__ attn, const float* __restrict__ temp,
                               bf16_t* __restrict__ out)
{
  int row = blockIdx.x * blockDim.x + threadIdx.x;  // BATCH*NHEADS*32
  if (row >= BATCH * NHEADS * 32) return;
  int h = (row >> 5) % NHEADS;
  const float* ip = attn + (long)row * 32;
  float t = temp[h];
  float vals[32], mx = -3.4e38f;
  #pragma unroll
  for (int j = 0; j < 32; ++j) { vals[j] = ip[j] * t; mx = fmaxf(mx, vals[j]); }
  float s = 0.f;
  #pragma unroll
  for (int j = 0; j < 32; ++j) { vals[j] = expf(vals[j] - mx); s += vals[j]; }
  float inv = 1.f / s;
  bf16_t* op = out + (long)row * 32;
  #pragma unroll
  for (int j = 0; j < 32; ++j) op[j] = f2bf(vals[j] * inv);
}

// g = gelu(x1) * fused  (bf16 NHWC), x1 = xdw channels [0,576)
__global__ void gelu_mul_kernel(const float* __restrict__ xdw, const float* __restrict__ fused,
                                bf16_t* __restrict__ out)
{
  long idx = (long)blockIdx.x * blockDim.x + threadIdx.x;  // (b,c,p), d=576
  long total = (long)BATCH * HIDDIM * PIX;
  if (idx >= total) return;
  int p = (int)(idx % PIX);
  long bc = idx / PIX;
  int c = (int)(bc % HIDDIM), bi = (int)(bc / HIDDIM);
  float x1 = xdw[(long)bi * 1152 * PIX + (long)c * PIX + p];
  float f = fused[idx];
  float g = 0.5f * x1 * (1.f + erff(x1 * 0.70710678118654752f));
  out[((long)bi * PIX + p) * HIDDIM + c] = f2bf(g * f);
}

// ------------------------------ host side ----------------------------------

static void launch_gemm(hipStream_t s,
    const bf16_t* A, const bf16_t* B, float* C, bf16_t* Cbf,
    int M, int N, int K, int lda, int ldb, int ldc, int ldcbf, int batches,
    long aO, long aI, int aM, long bO, long bI, int bM,
    long cO, long cI, int cM, long cbO, long cbI, int cbM, int acc)
{
  dim3 grid((N + 127) / 128, (M + 63) / 64, batches);
  gemm_wmma_bf16<<<grid, 256, 0, s>>>(A, B, C, Cbf, M, N, K, lda, ldb, ldc, ldcbf,
      aO, aI, aM, bO, bI, bM, cO, cI, cM, cbO, cbI, cbM, acc);
}

extern "C" void kernel_launch(void* const* d_in, const int* in_sizes, int n_in,
                              void* d_out, int out_size, void* d_ws, size_t ws_size,
                              hipStream_t stream) {
  (void)in_sizes; (void)n_in; (void)out_size; (void)ws_size;
  const float* x_in    = (const float*)d_in[0];
  const float* feat_in = (const float*)d_in[1];
  const float* nf_w = (const float*)d_in[2];  const float* nf_b = (const float*)d_in[3];
  const float* n1_w = (const float*)d_in[4];  const float* n1_b = (const float*)d_in[5];
  const float* n2_w = (const float*)d_in[6];  const float* n2_b = (const float*)d_in[7];
  const float* a_cw    = (const float*)d_in[8];
  const float* a_fw    = (const float*)d_in[9];
  const float* a_mw    = (const float*)d_in[10];
  const float* a_qkvw  = (const float*)d_in[11];
  const float* a_qkvdw = (const float*)d_in[12];
  const float* a_temp  = (const float*)d_in[13];
  const float* a_projw = (const float*)d_in[14];
  const float* f_pmw   = (const float*)d_in[15];
  const float* f_piw   = (const float*)d_in[16];
  const float* f_dww   = (const float*)d_in[17];
  const float* f_cwp   = (const float*)d_in[18];
  const float* f_fwp   = (const float*)d_in[19];
  const float* f_mwp   = (const float*)d_in[20];
  const float* f_pow   = (const float*)d_in[21];

  char* ws = (char*)d_ws;
  size_t cur = 0;
  auto alloc = [&](long elems, size_t esz) -> void* {
    void* p = ws + cur;
    cur += ((size_t)elems * esz + 255) & ~(size_t)255;
    return p;
  };
  auto cvtw = [&](const float* src, long n) -> bf16_t* {
    bf16_t* dst = (bf16_t*)alloc(n, 2);
    f32_to_bf16_kernel<<<(int)((n + 255) / 256), 256, 0, stream>>>(src, dst, n);
    return dst;
  };

  const long P = PIX;
  const long S192 = (long)BATCH * 192 * P;
  const long S384 = 2 * S192, S576 = 3 * S192, S1152 = 6 * S192;
  const long SP192 = 192 * P, SP384 = 384 * P, SP576 = 576 * P, SP1152 = 1152 * P;

  // ---- weights -> bf16 (persistent) ----
  bf16_t* w_acw  = cvtw(a_cw,   384L * 384);
  bf16_t* w_afw  = cvtw(a_fw,   192L * 384);
  bf16_t* w_amw  = cvtw(a_mw,   192L * 192);
  bf16_t* w_aqkv = cvtw(a_qkvw, 576L * 192);
  bf16_t* w_aprj = cvtw(a_projw,192L * 192);
  bf16_t* w_fpm  = cvtw(f_pmw,  576L * 192);
  bf16_t* w_fpi  = cvtw(f_piw,  1152L * 192);
  bf16_t* w_fcw  = cvtw(f_cwp,  1152L * 1152);
  bf16_t* w_ffw  = cvtw(f_fwp,  576L * 1152);
  bf16_t* w_fmw  = cvtw(f_mwp,  576L * 576);
  bf16_t* w_fpo  = cvtw(f_pow,  192L * 576);

  // ---- persistent activations ----
  bf16_t* feat_bf  = (bf16_t*)alloc(S192, 2);
  float*  x2_f32   = (float*)alloc(S192, 4);
  float*  attn_f32 = (float*)alloc((long)BATCH * NHEADS * 32 * 32, 4);
  bf16_t* attnsm   = (bf16_t*)alloc((long)BATCH * NHEADS * 32 * 32, 2);
  size_t mark = cur;

  // ---- phase A (attention) buffers ----
  float*  xln_f32 = (float*)alloc(S192, 4);   bf16_t* xln_bf = (bf16_t*)alloc(S192, 2);
  float*  mt_f32  = (float*)alloc(S192, 4);   bf16_t* mt_bf  = (bf16_t*)alloc(S192, 2);
  float*  cc_f32  = (float*)alloc(S384, 4);
  bf16_t* out1_bf = (bf16_t*)alloc(S192, 2);  bf16_t* out2_bf = (bf16_t*)alloc(S192, 2);
  float*  dx_f32  = (float*)alloc(S192, 4);   bf16_t* dx_bf  = (bf16_t*)alloc(S192, 2);
  float*  qkvpre  = (float*)alloc(S576, 4);
  float*  qkv_f32 = (float*)alloc(S576, 4);
  bf16_t* qn_bf   = (bf16_t*)alloc(S192, 2);  bf16_t* kn_bf  = (bf16_t*)alloc(S192, 2);
  bf16_t* v_bf    = (bf16_t*)alloc(S192, 2);
  bf16_t* attno_bf= (bf16_t*)alloc(S192, 2);
  float*  y_f32   = (float*)alloc(S192, 4);

  const int BP = (int)((BATCH * P + 255) / 256);
  auto g1 = [](long n) { return (int)((n + 255) / 256); };

  // LayerNorms
  layernorm_kernel<<<BP, 256, 0, stream>>>(feat_in, nf_w, nf_b, nullptr, feat_bf, 192);
  layernorm_kernel<<<BP, 256, 0, stream>>>(x_in, n1_w, n1_b, xln_f32, xln_bf, 192);

  // mt = mask conv(feat)
  launch_gemm(stream, w_amw, feat_bf, mt_f32, mt_bf, 192, (int)P, 192,
              192, 192, (int)P, 192, BATCH,
              0,0,1, SP192,0,1, SP192,0,1, SP192,0,1, 0);
  // cc = concat conv:  W[:, :192]*xln + W[:, 192:]*mt
  launch_gemm(stream, w_acw, xln_bf, cc_f32, nullptr, 384, (int)P, 192,
              384, 192, (int)P, 0, BATCH, 0,0,1, SP192,0,1, SP384,0,1, 0,0,1, 0);
  launch_gemm(stream, w_acw + 192, mt_bf, cc_f32, nullptr, 384, (int)P, 192,
              384, 192, (int)P, 0, BATCH, 0,0,1, SP192,0,1, SP384,0,1, 0,0,1, 1);
  // DDF elementwise
  ddf_elemwise_kernel<<<g1(S192), 256, 0, stream>>>(cc_f32, xln_f32, SP192, 0, mt_f32,
                                                    out1_bf, out2_bf, 192);
  // dx = fusion conv + xln
  launch_gemm(stream, w_afw, out1_bf, dx_f32, nullptr, 192, (int)P, 192,
              384, 192, (int)P, 0, BATCH, 0,0,1, SP192,0,1, SP192,0,1, 0,0,1, 0);
  launch_gemm(stream, w_afw + 192, out2_bf, dx_f32, nullptr, 192, (int)P, 192,
              384, 192, (int)P, 0, BATCH, 0,0,1, SP192,0,1, SP192,0,1, 0,0,1, 1);
  residual_kernel<<<g1(S192), 256, 0, stream>>>(dx_f32, xln_f32, SP192, 0, dx_f32, dx_bf, 192);
  // qkv = dw3x3(conv1x1(dx))
  launch_gemm(stream, w_aqkv, dx_bf, qkvpre, nullptr, 576, (int)P, 192,
              192, 192, (int)P, 0, BATCH, 0,0,1, SP192,0,1, SP576,0,1, 0,0,1, 0);
  dwconv3_kernel<<<g1(S576), 256, 0, stream>>>(qkvpre, a_qkvdw, qkv_f32, 576);
  // q/k l2-normalized (NCHW bf16), v -> NHWC bf16
  l2norm_rows_kernel<<<BATCH * 192, 256, 0, stream>>>(qkv_f32, SP576, 192, qn_bf);
  l2norm_rows_kernel<<<BATCH * 192, 256, 0, stream>>>(qkv_f32 + 192 * P, SP576, 192, kn_bf);
  nchw_to_nhwc_bf16_kernel<<<g1(S192), 256, 0, stream>>>(qkv_f32, v_bf, 192, 576, 384);
  // attn = q . k^T   (per b,head: 32x32, K=P)
  launch_gemm(stream, qn_bf, kn_bf, attn_f32, nullptr, 32, 32, (int)P,
              (int)P, (int)P, 32, 0, BATCH * NHEADS,
              32 * P, 0, 1, 32 * P, 0, 1, 1024, 0, 1, 0, 0, 1, 0);
  softmax_kernel<<<2, 256, 0, stream>>>(attn_f32, a_temp, attnsm);
  // out = attn . v  -> NHWC bf16 directly
  launch_gemm(stream, attnsm, v_bf, nullptr, attno_bf, 32, (int)P, 32,
              32, 192, 0, 192, BATCH * NHEADS,
              1024, 0, 1, P * 192, 32, NHEADS, 0, 0, 1, P * 192, 32, NHEADS, 0);
  // y = proj(out)
  launch_gemm(stream, w_aprj, attno_bf, y_f32, nullptr, 192, (int)P, 192,
              192, 192, (int)P, 0, BATCH, 0,0,1, SP192,0,1, SP192,0,1, 0,0,1, 0);
  // x2 = x + y
  residual_kernel<<<g1(S192), 256, 0, stream>>>(y_f32, x_in, SP192, 0, x2_f32, nullptr, 192);

  // ---- phase B (FFN) buffers (reuse phase-A region; stream order serializes) ----
  cur = mark;
  bf16_t* x2ln_bf = (bf16_t*)alloc(S192, 2);
  float*  xin_f32 = (float*)alloc(S1152, 4);
  float*  xdw_f32 = (float*)alloc(S1152, 4);
  bf16_t* fm_bf   = (bf16_t*)alloc(S576, 2);
  bf16_t* x2s_bf  = (bf16_t*)alloc(S576, 2);
  float*  mt2_f32 = (float*)alloc(S576, 4);   bf16_t* mt2_bf = (bf16_t*)alloc(S576, 2);
  float*  cc2_f32 = (float*)alloc(S1152, 4);
  bf16_t* o1_bf   = (bf16_t*)alloc(S576, 2);  bf16_t* o2_bf  = (bf16_t*)alloc(S576, 2);
  float*  fused   = (float*)alloc(S576, 4);
  bf16_t* g_bf    = (bf16_t*)alloc(S576, 2);
  float*  ffo_f32 = (float*)alloc(S192, 4);

  layernorm_kernel<<<BP, 256, 0, stream>>>(x2_f32, n2_w, n2_b, nullptr, x2ln_bf, 192);
  // xin = proj_in(x2ln)  [1152 ch]
  launch_gemm(stream, w_fpi, x2ln_bf, xin_f32, nullptr, 1152, (int)P, 192,
              192, 192, (int)P, 0, BATCH, 0,0,1, SP192,0,1, SP1152,0,1, 0,0,1, 0);
  // fm = proj_mask(feat) [576 ch] -> bf16 only
  launch_gemm(stream, w_fpm, feat_bf, nullptr, fm_bf, 576, (int)P, 192,
              192, 192, 0, 576, BATCH, 0,0,1, SP192,0,1, 0,0,1, P * 576, 0, 1, 0);
  // depthwise, split
  dwconv3_kernel<<<g1(S1152), 256, 0, stream>>>(xin_f32, f_dww, xdw_f32, 1152);
  nchw_to_nhwc_bf16_kernel<<<g1(S576), 256, 0, stream>>>(xdw_f32, x2s_bf, 576, 1152, 576);
  // mt2 = mask conv(fm)
  launch_gemm(stream, w_fmw, fm_bf, mt2_f32, mt2_bf, 576, (int)P, 576,
              576, 576, (int)P, 576, BATCH,
              0,0,1, P * 576, 0, 1, SP576, 0, 1, P * 576, 0, 1, 0);
  // cc2 = concat conv
  launch_gemm(stream, w_fcw, x2s_bf, cc2_f32, nullptr, 1152, (int)P, 576,
              1152, 576, (int)P, 0, BATCH, 0,0,1, P*576,0,1, SP1152,0,1, 0,0,1, 0);
  launch_gemm(stream, w_fcw + 576, mt2_bf, cc2_f32, nullptr, 1152, (int)P, 576,
              1152, 576, (int)P, 0, BATCH, 0,0,1, P*576,0,1, SP1152,0,1, 0,0,1, 1);
  ddf_elemwise_kernel<<<g1(S576), 256, 0, stream>>>(cc2_f32, xdw_f32, SP1152, 576, mt2_f32,
                                                    o1_bf, o2_bf, 576);
  // fused = fusion conv + x2s
  launch_gemm(stream, w_ffw, o1_bf, fused, nullptr, 576, (int)P, 576,
              1152, 576, (int)P, 0, BATCH, 0,0,1, P*576,0,1, SP576,0,1, 0,0,1, 0);
  launch_gemm(stream, w_ffw + 576, o2_bf, fused, nullptr, 576, (int)P, 576,
              1152, 576, (int)P, 0, BATCH, 0,0,1, P*576,0,1, SP576,0,1, 0,0,1, 1);
  residual_kernel<<<g1(S576), 256, 0, stream>>>(fused, xdw_f32, SP1152, 576, fused, nullptr, 576);
  // g = gelu(x1) * fused -> bf16 NHWC
  gelu_mul_kernel<<<g1(S576), 256, 0, stream>>>(xdw_f32, fused, g_bf);
  // ffo = proj_out(g)
  launch_gemm(stream, w_fpo, g_bf, ffo_f32, nullptr, 192, (int)P, 576,
              576, 576, (int)P, 0, BATCH, 0,0,1, P*576,0,1, SP192,0,1, 0,0,1, 0);
  // out = x2 + ffo
  residual_kernel<<<g1(S192), 256, 0, stream>>>(x2_f32, ffo_f32, SP192, 0,
                                                (float*)d_out, nullptr, 192);
}
